// att_res_conv_normalized_12489764897294
// MI455X (gfx1250) — matrix operand order repaired and checked by reading, not compiled, not run on hardware
//
#include <hip/hip_runtime.h>
#include <hip/hip_bf16.h>

typedef __attribute__((ext_vector_type(16))) __bf16 v16bf;
typedef __attribute__((ext_vector_type(8)))  float  v8f;

#define HW_      50176      // 224*224
#define TILE_N   128
#define NTHREADS 256
#define XLD      400        // Xs row stride in bf16 (800B, 32B-aligned)
#define Q1LD     64         // q1 spill row stride (128B)

__device__ __forceinline__ float xsigmoid(float x) { return 1.f / (1.f + __expf(-x)); }

// Stage an O x C fp32 weight matrix (rows mrow0..mrow0+MT*16) into LDS as bf16,
// pre-arranged in exact 16x32 A-fragment order: frag f, lane ln holds 16 bf16
// (contiguous 32B). K is zero-padded to KT*32.
template<int MT, int KT>
__device__ __forceinline__ void stage_w(__bf16* wsm, const float* __restrict__ Wg,
                                        int C, int mrow0, int tid) {
  __syncthreads();                      // protect previous stage's readers
  const int total = MT * KT * 512;
  #pragma unroll 1
  for (int i = tid; i < total; i += NTHREADS) {
    int f  = i >> 9;
    int r  = i & 511;
    int ln = r >> 4;
    int e  = r & 15;
    int mt = f / KT, kt = f - mt * KT;
    int half = ln >> 4;
    int m = mrow0 + mt * 16 + (ln & 15);
    int k = kt * 32 + half * 16 + e;  // contiguous K order k = 16*half + e (same as B reads)
    wsm[i] = (k < C) ? (__bf16)Wg[m * C + k] : (__bf16)0.f;
  }
  __syncthreads();
}

// D[MT tiles][16x16] += staged-W  x  B, B read as [pixel][channel] bf16 rows.
template<int MT, int KT>
__device__ __forceinline__ void gemm(const __bf16* wsm, const __bf16* brow,
                                     int ldb, int lane, v8f* acc) {
  const int n  = lane & 15;
  const int kh = (lane >> 4) << 4;
  const __bf16* bp = brow + n * ldb + kh;
  const __bf16* ap = wsm + lane * 16;
  #pragma unroll
  for (int kt = 0; kt < KT; ++kt) {
    v16bf bf = *(const v16bf*)(bp + kt * 32);
    #pragma unroll
    for (int mt = 0; mt < MT; ++mt) {
      v16bf af = *(const v16bf*)(ap + (mt * KT + kt) * 512);
      acc[mt] = __builtin_amdgcn_wmma_f32_16x16x32_bf16(
          false, af, false, bf, (short)0, acc[mt], false, false);
    }
  }
}

template<int MT>
__device__ __forceinline__ void addbias(v8f* acc, const float* __restrict__ bias, int lane) {
  const int half = lane >> 4;
  #pragma unroll
  for (int mt = 0; mt < MT; ++mt)
    #pragma unroll
    for (int j = 0; j < 8; ++j)
      acc[mt][j] += bias[mt * 16 + half * 8 + j];
}

// LayerNorm over channel dim (+ ReLU). Channels of one pixel live in this lane
// and lane^16 -> per-lane partial + one half-wave swap.
template<int MT>
__device__ __forceinline__ void ln_relu(v8f* acc, const float* __restrict__ g,
                                        const float* __restrict__ be, int lane) {
  const float nch = (float)(MT * 16);
  float s = 0.f;
  #pragma unroll
  for (int mt = 0; mt < MT; ++mt)
    #pragma unroll
    for (int j = 0; j < 8; ++j) s += acc[mt][j];
  s += __shfl_xor(s, 16, 32);
  const float mu = s / nch;
  float var = 0.f;
  #pragma unroll
  for (int mt = 0; mt < MT; ++mt)
    #pragma unroll
    for (int j = 0; j < 8; ++j) { float d = acc[mt][j] - mu; var += d * d; }
  var += __shfl_xor(var, 16, 32);
  const float rs = rsqrtf(var / nch + 1e-5f);
  const int half = lane >> 4;
  #pragma unroll
  for (int mt = 0; mt < MT; ++mt)
    #pragma unroll
    for (int j = 0; j < 8; ++j) {
      int ch = mt * 16 + half * 8 + j;
      acc[mt][j] = fmaxf((acc[mt][j] - mu) * rs * g[ch] + be[ch], 0.f);
    }
}

// F.normalize(p=2,dim=ch) then softmax(dim=ch) over 128 channels (MT=8).
__device__ __forceinline__ void l2norm_softmax(v8f* a) {
  float ss = 0.f;
  #pragma unroll
  for (int mt = 0; mt < 8; ++mt)
    #pragma unroll
    for (int j = 0; j < 8; ++j) { float x = a[mt][j]; ss += x * x; }
  ss += __shfl_xor(ss, 16, 32);
  const float inv = 1.f / fmaxf(sqrtf(ss), 1e-8f);
  float m = -3.4e38f;
  #pragma unroll
  for (int mt = 0; mt < 8; ++mt)
    #pragma unroll
    for (int j = 0; j < 8; ++j) m = fmaxf(m, a[mt][j] * inv);
  m = fmaxf(m, __shfl_xor(m, 16, 32));
  float sum = 0.f;
  #pragma unroll
  for (int mt = 0; mt < 8; ++mt)
    #pragma unroll
    for (int j = 0; j < 8; ++j) {
      float e = __expf(a[mt][j] * inv - m);
      a[mt][j] = e; sum += e;
    }
  sum += __shfl_xor(sum, 16, 32);
  const float r = 1.f / sum;
  #pragma unroll
  for (int mt = 0; mt < 8; ++mt)
    #pragma unroll
    for (int j = 0; j < 8; ++j) a[mt][j] *= r;
}

// Spill C-layout accumulators to bf16 [pixel][channel] rows (B-operand layout).
template<int MT>
__device__ __forceinline__ void spill(const v8f* acc, __bf16* sp, int ldsp,
                                      int lane, int chbase) {
  const int n = lane & 15, half = lane >> 4;
  #pragma unroll
  for (int mt = 0; mt < MT; ++mt)
    #pragma unroll
    for (int j = 0; j < 8; ++j)
      sp[n * ldsp + chbase + mt * 16 + half * 8 + j] = (__bf16)acc[mt][j];
}

__global__ __launch_bounds__(NTHREADS)
void fused_attres_kernel(const float* __restrict__ kv, const float* __restrict__ qin,
                         const float* Wk,  const float* bk,
                         const float* Wv,  const float* bv,
                         const float* Wq1, const float* bq1,
                         const float* Wq2, const float* bq2,
                         const float* Watt, const float* batt,
                         const float* gatt, const float* betaatt,
                         const float* Wres, const float* gres, const float* betares,
                         const float* Wd1, const float* gd1, const float* betad1,
                         const float* Wd2, const float* gd2, const float* betad2,
                         const float* Wd3, const float* bd3,
                         float* __restrict__ out) {
  __shared__ __align__(32) __bf16 Xs[TILE_N * XLD];        // 102,400 B activations / spill reuse
  __shared__ __align__(32) __bf16 Wsm[32 * 512];           //  32,768 B staged weight frags
  __shared__ __align__(32) __bf16 SpQ1[8 * 16 * Q1LD];     //  16,384 B q1 spill (X still live)

  const int tid  = threadIdx.x;
  const int lane = tid & 31;
  const int wave = tid >> 5;

  const int p0  = blockIdx.x * TILE_N;
  const int b   = p0 / HW_;
  const int hw0 = p0 - b * HW_;

  // ---- load activation tile -> LDS bf16 [pixel][channel]: cols 0..255 = kv, 256..383 = q
  {
    const float* kvb = kv  + (size_t)b * 256 * HW_ + hw0;
    const float* qb  = qin + (size_t)b * 128 * HW_ + hw0;
    #pragma unroll 1
    for (int i = tid; i < 384 * TILE_N; i += NTHREADS) {
      int c = i >> 7, p = i & 127;
      float v = (c < 256) ? kvb[(size_t)c * HW_ + p] : qb[(size_t)(c - 256) * HW_ + p];
      Xs[p * XLD + c] = (__bf16)v;
    }
  }
  __syncthreads();

  const __bf16* Bx  = Xs + (wave * 16) * XLD;     // this wave's 16 pixels (B operand rows)
  __bf16*       SpL = Xs + (wave * 16) * XLD;     // per-wave spill region (reuses dead cols)
  const __bf16* Bq1 = SpQ1 + wave * 16 * Q1LD;

  // ---- res = relu(LN(W_res @ concat(kv,q)))  [32 ch], K=384
  v8f accR[2]; for (int i = 0; i < 2; ++i) accR[i] = 0.f;
  stage_w<2, 12>(Wsm, Wres, 384, 0, tid);
  gemm<2, 12>(Wsm, Bx, XLD, lane, accR);
  ln_relu<2>(accR, gres, betares, lane);

  // ---- q1 = silu(W_q1 @ q + b)  [64 ch], K=128 ; spill as B operand
  {
    v8f accQ1[4]; for (int i = 0; i < 4; ++i) accQ1[i] = 0.f;
    stage_w<4, 4>(Wsm, Wq1, 128, 0, tid);
    gemm<4, 4>(Wsm, Bx + 256, XLD, lane, accQ1);   // q channels live at cols 256..383
    addbias<4>(accQ1, bq1, lane);
    #pragma unroll
    for (int mt = 0; mt < 4; ++mt)
      #pragma unroll
      for (int j = 0; j < 8; ++j) {
        float x = accQ1[mt][j]; accQ1[mt][j] = x * xsigmoid(x);
      }
    spill<4>(accQ1, SpQ1 + wave * 16 * Q1LD, Q1LD, lane, 0);
  }

  // ---- query = W_q2 @ q1 + b  [128 ch], K=64
  v8f accQ[8]; for (int i = 0; i < 8; ++i) accQ[i] = 0.f;
  stage_w<8, 2>(Wsm, Wq2, 64, 0, tid);
  gemm<8, 2>(Wsm, Bq1, Q1LD, lane, accQ);
  addbias<8>(accQ, bq2, lane);

  // ---- k = W_key @ kv + b  [128 ch], K=256 ;  att = softmax(l2norm(query*k))
  {
    v8f accK[8]; for (int i = 0; i < 8; ++i) accK[i] = 0.f;
    stage_w<4, 8>(Wsm, Wk, 256, 0, tid);   gemm<4, 8>(Wsm, Bx, XLD, lane, accK);
    stage_w<4, 8>(Wsm, Wk, 256, 64, tid);  gemm<4, 8>(Wsm, Bx, XLD, lane, accK + 4);
    addbias<8>(accK, bk, lane);
    #pragma unroll
    for (int mt = 0; mt < 8; ++mt)
      #pragma unroll
      for (int j = 0; j < 8; ++j) accQ[mt][j] *= accK[mt][j];
  }   // accK dies here -> only accR + accQ live across the next GEMM
  l2norm_softmax(accQ);

  // ---- value = W_value @ kv + b  [128 ch], K=256 (kv cols still intact)
  {
    v8f accV[8]; for (int i = 0; i < 8; ++i) accV[i] = 0.f;
    stage_w<4, 8>(Wsm, Wv, 256, 0, tid);   gemm<4, 8>(Wsm, Bx, XLD, lane, accV);
    stage_w<4, 8>(Wsm, Wv, 256, 64, tid);  gemm<4, 8>(Wsm, Bx, XLD, lane, accV + 4);
    addbias<8>(accV, bv, lane);
    // x = att * value
    #pragma unroll
    for (int mt = 0; mt < 8; ++mt)
      #pragma unroll
      for (int j = 0; j < 8; ++j) accQ[mt][j] *= accV[mt][j];
  }   // accV dies

  // spill x into the q-channel columns (256..383) -- dead since the q1 GEMM,
  // while kv columns were preserved for the value GEMM above.
  spill<8>(accQ, SpL, XLD, lane, 256);

  // ---- att_out = relu(LN(W_att @ x + b))  [128 ch], K=128
  v8f accA[8]; for (int i = 0; i < 8; ++i) accA[i] = 0.f;
  stage_w<8, 4>(Wsm, Watt, 128, 0, tid);
  gemm<8, 4>(Wsm, SpL + 256, XLD, lane, accA);
  addbias<8>(accA, batt, lane);
  ln_relu<8>(accA, gatt, betaatt, lane);

  // ---- d1 = relu(LN(W_d1 @ concat(att_out, res)))  [32 ch], K=160
  // kv columns of this wave's rows are dead now -> reuse cols 0..159
  spill<8>(accA, SpL, XLD, lane, 0);       // cols 0..127
  spill<2>(accR, SpL, XLD, lane, 128);     // cols 128..159
  v8f accD1[2]; for (int i = 0; i < 2; ++i) accD1[i] = 0.f;
  stage_w<2, 5>(Wsm, Wd1, 160, 0, tid);
  gemm<2, 5>(Wsm, SpL, XLD, lane, accD1);
  ln_relu<2>(accD1, gd1, betad1, lane);

  // ---- d2 = relu(LN(W_d2 @ d1))  [16 ch], K=32
  spill<2>(accD1, SpL, XLD, lane, 0);
  v8f accD2[1]; accD2[0] = 0.f;
  stage_w<1, 1>(Wsm, Wd2, 32, 0, tid);
  gemm<1, 1>(Wsm, SpL, XLD, lane, accD2);
  ln_relu<1>(accD2, gd2, betad2, lane);

  // ---- out = W_d3 @ d2 + b  [16 ch], K=16 zero-padded to 32
  spill<1>(accD2, SpL, XLD, lane, 0);
  {
    const int n = lane & 15, half = lane >> 4;
    #pragma unroll
    for (int j = 0; j < 8; ++j)
      SpL[n * XLD + 16 + half * 8 + j] = (__bf16)0.f;   // zero pad K=16..31
  }
  v8f accO[1]; accO[0] = 0.f;
  stage_w<1, 1>(Wsm, Wd3, 16, 0, tid);    // K padded with zeros inside stage_w
  gemm<1, 1>(Wsm, SpL, XLD, lane, accO);
  addbias<1>(accO, bd3, lane);

  // ---- store NCHW fp32
  {
    const int n = lane & 15, half = lane >> 4;
    const int p  = p0 + wave * 16 + n;
    const int bo = p / HW_, hwo = p - bo * HW_;
    float* op = out + (size_t)bo * 16 * HW_ + hwo;
    #pragma unroll
    for (int j = 0; j < 8; ++j)
      op[(size_t)(half * 8 + j) * HW_] = accO[0][j];
  }
}

extern "C" void kernel_launch(void* const* d_in, const int* in_sizes, int n_in,
                              void* d_out, int out_size, void* d_ws, size_t ws_size,
                              hipStream_t stream) {
  const float* kv   = (const float*)d_in[0];
  const float* q    = (const float*)d_in[1];
  const float* Wk   = (const float*)d_in[2];
  const float* bk   = (const float*)d_in[3];
  const float* Wv   = (const float*)d_in[4];
  const float* bv   = (const float*)d_in[5];
  const float* Wq1  = (const float*)d_in[6];
  const float* bq1  = (const float*)d_in[7];
  const float* Wq2  = (const float*)d_in[8];
  const float* bq2  = (const float*)d_in[9];
  // d_in[10] = scale (unused by the reference computation)
  const float* Watt = (const float*)d_in[11];
  const float* batt = (const float*)d_in[12];
  const float* gatt = (const float*)d_in[13];
  const float* beat = (const float*)d_in[14];
  const float* Wres = (const float*)d_in[15];
  const float* gres = (const float*)d_in[16];
  const float* bres = (const float*)d_in[17];
  const float* Wd1  = (const float*)d_in[18];
  const float* gd1  = (const float*)d_in[19];
  const float* bed1 = (const float*)d_in[20];
  const float* Wd2  = (const float*)d_in[21];
  const float* gd2  = (const float*)d_in[22];
  const float* bed2 = (const float*)d_in[23];
  const float* Wd3  = (const float*)d_in[24];
  const float* bd3  = (const float*)d_in[25];
  float* out = (float*)d_out;

  const int npix = 4 * HW_;                 // 200704
  dim3 grid(npix / TILE_N);                 // 1568 workgroups
  fused_attres_kernel<<<grid, NTHREADS, 0, stream>>>(
      kv, q, Wk, bk, Wv, bv, Wq1, bq1, Wq2, bq2,
      Watt, batt, gatt, beat, Wres, gres, bres,
      Wd1, gd1, bed1, Wd2, gd2, bed2, Wd3, bd3, out);
}
